// GraphPDHGNet_73778948210745
// MI455X (gfx1250) — matrix operand order repaired and checked by reading, not compiled, not run on hardware
//
#include <hip/hip_runtime.h>
#include <math.h>

// ---------------- CDNA5 WMMA types ----------------
typedef __attribute__((ext_vector_type(16))) __bf16 v16bf;
typedef __attribute__((ext_vector_type(8)))  float  v8f;

__device__ __forceinline__ v8f wmma_bf16(v16bf a, v16bf b, v8f c) {
  // D = A(16x32 bf16) * B(32x16 bf16) + C(16x16 f32)
  return __builtin_amdgcn_wmma_f32_16x16x32_bf16(
      /*neg_a=*/false, a, /*neg_b=*/false, b,
      /*c_mod=*/(short)0, c, /*reuse_a=*/false, /*reuse_b=*/false);
}

// ---------------- bf16 split helpers (hi + residual lo) ----------------
// Native casts lower to v_cvt_pk_bf16_f32 / v_cvt_f32_bf16 on gfx1250.
__device__ __forceinline__ void split2(float x, __bf16& hi, __bf16& lo) {
  hi = (__bf16)x;
  lo = (__bf16)(x - (float)hi);
}

__device__ __forceinline__ v16bf pack16(const __bf16 (&a)[16]) {
  union { __bf16 b[16]; v16bf v; } x;
#pragma unroll
  for (int i = 0; i < 16; ++i) x.b[i] = a[i];
  return x.v;
}

// Load a pre-swizzled B fragment: 32 lanes x 8 dwords contiguous (1KB), two b128s.
__device__ __forceinline__ v16bf load_frag(const unsigned* __restrict__ p32, int lane) {
  const uint4* p = (const uint4*)(p32 + lane * 8);
  union { uint4 q[2]; v16bf v; } x;
  x.q[0] = p[0];
  x.q[1] = p[1];
  return x.v;
}

__device__ __forceinline__ void atom_add_f(float* p, float v) {
  (void)__hip_atomic_fetch_add(p, v, __ATOMIC_RELAXED, __HIP_MEMORY_SCOPE_AGENT);
}

// A-layout K index for (vgpr v, lane-group g): pairs (k, k+1)
__device__ __forceinline__ int a_kidx(int kt, int v, int g) {
  return kt * 32 + ((v & 4) ? 16 : 0) + g * 8 + (v & 3) * 2;
}

// ---------------- prep kernels ----------------
__global__ void zero_f32(float* __restrict__ p, long long n) {
  long long i = (long long)blockIdx.x * blockDim.x + threadIdx.x;
  if (i < n) p[i] = 0.0f;
}

__global__ void count_deg(const int* __restrict__ dst, float* __restrict__ cnt, int E) {
  int i = blockIdx.x * blockDim.x + threadIdx.x;
  if (i < E) atom_add_f(&cnt[dst[i]], 1.0f);
}

__global__ void recip_deg(float* __restrict__ c, int N) {
  int i = blockIdx.x * blockDim.x + threadIdx.x;
  if (i < N) c[i] = 1.0f / fmaxf(c[i], 1.0f);
}

// Swizzle fp32 weights [L][K][96] into WMMA B fragments, split hi/lo planes.
// out layout: [((l*nkt + kt)*6 + nt)*2 + half][lane*8 + v] dwords.
__global__ void swz_weights(const float* __restrict__ W, unsigned* __restrict__ out,
                            int K, int total) {
  int t = blockIdx.x * blockDim.x + threadIdx.x;
  if (t >= total) return;
  int v    = t & 7;
  int lane = (t >> 3) & 31;
  int half = (t >> 8) & 1;
  int rest = t >> 9;
  int nt = rest % 6; rest /= 6;
  int nkt = K >> 5;
  int kt = rest % nkt;
  int l  = rest / nkt;
  int n = nt * 16 + (lane & 15);
  int k = kt * 32 + ((v & 4) ? 16 : 0) + (lane >> 4) * 8 + (v & 3) * 2;   // B: same K order as A (a_kidx)
  const float* Wl = W + (size_t)l * K * 96;
  float x0 = Wl[(size_t)k * 96 + n];
  float x1 = Wl[(size_t)(k + 1) * 96 + n];
  union { __bf16 b[2]; unsigned u; } ph, pl;
  split2(x0, ph.b[0], pl.b[0]);
  split2(x1, ph.b[1], pl.b[1]);
  out[t] = half ? pl.u : ph.u;
}

// ---------------- edge kernel: e' = proj_{||.||<=w}(e@Weu + b + (h_s - h_d)@Wea + b); agg += e' ----------------
__global__ void __launch_bounds__(128)
edge_kernel(const float* __restrict__ e_in, float* __restrict__ e_out,
            const float* __restrict__ h,
            const int* __restrict__ src, const int* __restrict__ dst,
            const float* __restrict__ w,
            const unsigned* __restrict__ Weu, const unsigned* __restrict__ Wea,
            const float* __restrict__ beu, const float* __restrict__ bea,
            float* __restrict__ agg, int E) {
  const int lane = threadIdx.x & 31;
  const int wave = blockIdx.x * (blockDim.x >> 5) + (threadIdx.x >> 5);
  const long long strip = (long long)wave * 16;
  if (strip >= E) return;
  const int M = lane & 15, g = lane >> 4, col = M;

  const long long erow = strip + M;
  const int si = src[erow], di = dst[erow];
  const float* ep = e_in + erow * 96;
  const float* hs = h + (long long)si * 96;
  const float* hd = h + (long long)di * 96;

  // A fragments (hi/lo) for e-row and for (h_src - h_dst)
  v16bf Aeh[3], Ael[3], Adh[3], Adl[3];
#pragma unroll
  for (int kt = 0; kt < 3; ++kt) {
    __bf16 aeh[16], ael[16], adh[16], adl[16];
#pragma unroll
    for (int v = 0; v < 8; ++v) {
      const int k = a_kidx(kt, v, g);
      float2 ev = *(const float2*)(ep + k);
      float2 av = *(const float2*)(hs + k);
      float2 bv = *(const float2*)(hd + k);
      split2(ev.x, aeh[2 * v], ael[2 * v]);
      split2(ev.y, aeh[2 * v + 1], ael[2 * v + 1]);
      split2(av.x - bv.x, adh[2 * v], adl[2 * v]);
      split2(av.y - bv.y, adh[2 * v + 1], adl[2 * v + 1]);
    }
    Aeh[kt] = pack16(aeh); Ael[kt] = pack16(ael);
    Adh[kt] = pack16(adh); Adl[kt] = pack16(adl);
  }

  v8f acc[6];
  float rowsq[8];
#pragma unroll
  for (int r = 0; r < 8; ++r) rowsq[r] = 0.0f;

#pragma unroll
  for (int nt = 0; nt < 6; ++nt) {
    const float bias = beu[nt * 16 + col] + bea[nt * 16 + col];
    v8f c;
#pragma unroll
    for (int r = 0; r < 8; ++r) c[r] = bias;
#pragma unroll
    for (int kt = 0; kt < 3; ++kt) {
      const unsigned* fb = Weu + (kt * 6 + nt) * 512;
      v16bf Bh = load_frag(fb, lane);
      v16bf Bl = load_frag(fb + 256, lane);
      c = wmma_bf16(Aeh[kt], Bh, c);
      c = wmma_bf16(Aeh[kt], Bl, c);
      c = wmma_bf16(Ael[kt], Bh, c);
      fb = Wea + (kt * 6 + nt) * 512;
      Bh = load_frag(fb, lane);
      Bl = load_frag(fb + 256, lane);
      c = wmma_bf16(Adh[kt], Bh, c);
      c = wmma_bf16(Adh[kt], Bl, c);
      c = wmma_bf16(Adl[kt], Bh, c);
    }
    acc[nt] = c;
    // partial row sum-of-squares: reduce across the 16 lanes of this half-wave
#pragma unroll
    for (int r = 0; r < 8; ++r) {
      float t = c[r] * c[r];
      t += __shfl_xor(t, 1);
      t += __shfl_xor(t, 2);
      t += __shfl_xor(t, 4);
      t += __shfl_xor(t, 8);
      rowsq[r] += t;
    }
  }

  // per-row L2-ball projection scale; row M' = r + 8*g
  float scl[8];
  int drow[8];
#pragma unroll
  for (int r = 0; r < 8; ++r) {
    const long long er = strip + r + 8 * g;
    const float rad = w[er];                       // LAM = 1.0
    const float nrm = fmaxf(sqrtf(rowsq[r]), 1e-8f);
    scl[r] = fminf(1.0f, rad / nrm);
    drow[r] = dst[er];
  }

#pragma unroll
  for (int nt = 0; nt < 6; ++nt) {
#pragma unroll
    for (int r = 0; r < 8; ++r) {
      const float val = acc[nt][r] * scl[r];
      const long long er = strip + r + 8 * g;
      e_out[er * 96 + nt * 16 + col] = val;
      atom_add_f(&agg[(long long)drow[r] * 96 + nt * 16 + col], val);
    }
  }
}

// ---------------- node kernel: h' = silu([h | agg/deg] @ Wn1 + b) @ Wn2 + b ----------------
__global__ void __launch_bounds__(128)
node_kernel(const float* __restrict__ h_in, const float* __restrict__ agg,
            const float* __restrict__ rden,
            const unsigned* __restrict__ Wn1, const unsigned* __restrict__ Wn2,
            const float* __restrict__ bn1, const float* __restrict__ bn2,
            float* __restrict__ h_out, int N) {
  __shared__ float lds[4][16][97];   // per-wave 16x96 hidden tile (padded)
  const int lane = threadIdx.x & 31;
  const int wv = threadIdx.x >> 5;
  const int wave = blockIdx.x * 4 + wv;
  const long long strip = (long long)wave * 16;
  if (strip >= N) return;
  const int M = lane & 15, g = lane >> 4, col = M;
  const long long nrow = strip + M;
  const float* hp = h_in + nrow * 96;
  const float* ap = agg + nrow * 96;
  const float rd = rden[nrow];

  // x = [h | agg * rden], K = 192 -> 6 A fragments (hi/lo)
  v16bf Axh[6], Axl[6];
#pragma unroll
  for (int kt = 0; kt < 6; ++kt) {
    __bf16 uh[16], ul[16];
#pragma unroll
    for (int v = 0; v < 8; ++v) {
      const int k = a_kidx(kt, v, g);
      float x0, x1;
      if (k < 96) {
        float2 t = *(const float2*)(hp + k);
        x0 = t.x; x1 = t.y;
      } else {
        float2 t = *(const float2*)(ap + (k - 96));
        x0 = t.x * rd; x1 = t.y * rd;
      }
      split2(x0, uh[2 * v], ul[2 * v]);
      split2(x1, uh[2 * v + 1], ul[2 * v + 1]);
    }
    Axh[kt] = pack16(uh);
    Axl[kt] = pack16(ul);
  }

  // GEMM1 + SiLU, stash hidden tile in LDS (same-wave DS ordering)
#pragma unroll
  for (int nt = 0; nt < 6; ++nt) {
    const float bias = bn1[nt * 16 + col];
    v8f c;
#pragma unroll
    for (int r = 0; r < 8; ++r) c[r] = bias;
#pragma unroll
    for (int kt = 0; kt < 6; ++kt) {
      const unsigned* fb = Wn1 + (kt * 6 + nt) * 512;
      v16bf Bh = load_frag(fb, lane);
      v16bf Bl = load_frag(fb + 256, lane);
      c = wmma_bf16(Axh[kt], Bh, c);
      c = wmma_bf16(Axh[kt], Bl, c);
      c = wmma_bf16(Axl[kt], Bh, c);
    }
#pragma unroll
    for (int r = 0; r < 8; ++r) {
      float x = c[r];
      lds[wv][r + 8 * g][nt * 16 + col] = x / (1.0f + __expf(-x));
    }
  }

  // Re-layout hidden C-tile -> A fragments via LDS
  v16bf Ahh[3], Ahl[3];
#pragma unroll
  for (int kt = 0; kt < 3; ++kt) {
    __bf16 uh[16], ul[16];
#pragma unroll
    for (int v = 0; v < 8; ++v) {
      const int k = a_kidx(kt, v, g);
      float x0 = lds[wv][M][k];
      float x1 = lds[wv][M][k + 1];
      split2(x0, uh[2 * v], ul[2 * v]);
      split2(x1, uh[2 * v + 1], ul[2 * v + 1]);
    }
    Ahh[kt] = pack16(uh);
    Ahl[kt] = pack16(ul);
  }

  // GEMM2 -> h_out
#pragma unroll
  for (int nt = 0; nt < 6; ++nt) {
    const float bias = bn2[nt * 16 + col];
    v8f c;
#pragma unroll
    for (int r = 0; r < 8; ++r) c[r] = bias;
#pragma unroll
    for (int kt = 0; kt < 3; ++kt) {
      const unsigned* fb = Wn2 + (kt * 6 + nt) * 512;
      v16bf Bh = load_frag(fb, lane);
      v16bf Bl = load_frag(fb + 256, lane);
      c = wmma_bf16(Ahh[kt], Bh, c);
      c = wmma_bf16(Ahh[kt], Bl, c);
      c = wmma_bf16(Ahl[kt], Bh, c);
    }
#pragma unroll
    for (int r = 0; r < 8; ++r)
      h_out[(strip + r + 8 * g) * 96 + nt * 16 + col] = c[r];
  }
}

// ---------------- host launcher ----------------
extern "C" void kernel_launch(void* const* d_in, const int* in_sizes, int n_in,
                              void* d_out, int out_size, void* d_ws, size_t ws_size,
                              hipStream_t stream) {
  const float* h0  = (const float*)d_in[0];
  const float* e0  = (const float*)d_in[1];
  const float* w   = (const float*)d_in[2];
  const float* Weu = (const float*)d_in[3];
  const float* beu = (const float*)d_in[4];
  const float* Wea = (const float*)d_in[5];
  const float* bea = (const float*)d_in[6];
  const float* Wn1 = (const float*)d_in[7];
  const float* bn1 = (const float*)d_in[8];
  const float* Wn2 = (const float*)d_in[9];
  const float* bn2 = (const float*)d_in[10];
  const int* eidx  = (const int*)d_in[11];

  const int E = in_sizes[2];          // w has shape (E,)
  const int N = in_sizes[0] / 96;     // h has shape (N, 96)
  const int L = 4;
  const int* src = eidx;
  const int* dst = eidx + E;

  float* out_h = (float*)d_out;
  float* out_e = out_h + (size_t)N * 96;

  // workspace carve (all 256B aligned)
  char* base = (char*)d_ws;
  size_t off = 0;
  auto carve = [&](size_t bytes) -> void* {
    void* p = base + off;
    off += (bytes + 255) & ~(size_t)255;
    return p;
  };
  float* hA   = (float*)carve((size_t)N * 96 * 4);
  float* hB   = (float*)carve((size_t)N * 96 * 4);
  float* agg  = (float*)carve((size_t)N * 96 * 4);
  float* rden = (float*)carve((size_t)N * 4);
  const int edgeWdw = L * 3 * 6 * 512;   // dwords per [L] K=96 weight, split hi/lo
  const int n1Wdw   = L * 6 * 6 * 512;   // K=192
  unsigned* sWeu = (unsigned*)carve((size_t)edgeWdw * 4);
  unsigned* sWea = (unsigned*)carve((size_t)edgeWdw * 4);
  unsigned* sWn1 = (unsigned*)carve((size_t)n1Wdw * 4);
  unsigned* sWn2 = (unsigned*)carve((size_t)edgeWdw * 4);

  // degree -> 1/max(cnt,1)
  zero_f32<<<(N + 255) / 256, 256, 0, stream>>>(rden, N);
  count_deg<<<(E + 255) / 256, 256, 0, stream>>>(dst, rden, E);
  recip_deg<<<(N + 255) / 256, 256, 0, stream>>>(rden, N);

  // weight swizzle to WMMA-B split-bf16 fragments
  swz_weights<<<(edgeWdw + 255) / 256, 256, 0, stream>>>(Weu, sWeu, 96, edgeWdw);
  swz_weights<<<(edgeWdw + 255) / 256, 256, 0, stream>>>(Wea, sWea, 96, edgeWdw);
  swz_weights<<<(n1Wdw + 255) / 256, 256, 0, stream>>>(Wn1, sWn1, 192, n1Wdw);
  swz_weights<<<(edgeWdw + 255) / 256, 256, 0, stream>>>(Wn2, sWn2, 96, edgeWdw);

  const int eBlocks = ((E + 15) / 16 + 3) / 4;
  const int nBlocks = ((N + 15) / 16 + 3) / 4;
  const float* hcur = h0;
  const float* ecur = e0;
  for (int l = 0; l < L; ++l) {
    zero_f32<<<(int)(((long long)N * 96 + 255) / 256), 256, 0, stream>>>(agg, (long long)N * 96);
    edge_kernel<<<eBlocks, 128, 0, stream>>>(
        ecur, out_e, hcur, src, dst, w,
        sWeu + (size_t)l * 3 * 6 * 512, sWea + (size_t)l * 3 * 6 * 512,
        beu + l * 96, bea + l * 96, agg, E);
    ecur = out_e;
    float* hn = (l == L - 1) ? out_h : ((l & 1) ? hB : hA);
    node_kernel<<<nBlocks, 128, 0, stream>>>(
        hcur, agg, rden,
        sWn1 + (size_t)l * 6 * 6 * 512, sWn2 + (size_t)l * 3 * 6 * 512,
        bn1 + l * 96, bn2 + l * 96, hn, N);
    hcur = hn;
  }
}
